// SingleHeadAttention_23201413333083
// MI455X (gfx1250) — compile-verified, matrix operand order checked by reading, not run on hardware
//
#include <hip/hip_runtime.h>
#include <hip/hip_bf16.h>

// ---------------------------------------------------------------------------
// Single-head attention for MI455X (gfx1250, wave32, WMMA bf16 f32-acc).
// bf16 convert once; all matmuls via v_wmma_f32_16x16x32_bf16.
// P (4x2048x2048 bf16 = 32MB) is materialized: L2-resident (192MB L2).
// LDS tiles are filled with GLOBAL_LOAD_ASYNC_TO_LDS_B128 (ASYNCcnt path).
// ---------------------------------------------------------------------------

#define BATCH  4
#define TSEQ   2048
#define DMODEL 1024
#define BT     (BATCH * TSEQ)   // 8192

typedef __attribute__((ext_vector_type(16))) __bf16 v16bf;
typedef __attribute__((ext_vector_type(8)))  __bf16 v8bf;
typedef __attribute__((ext_vector_type(8)))  float  v8f;

__device__ __forceinline__ int lane_id() { return (int)(threadIdx.x & 31); }

// Async DMA 16 bytes global -> LDS (gfx1250 GLOBAL_LOAD_ASYNC_TO_LDS_B128).
// lds_off: LDS byte address (low 32 bits of generic pointer == LDS offset).
__device__ __forceinline__ void async_copy_b128(unsigned lds_off, const void* gptr) {
  asm volatile("global_load_async_to_lds_b128 %0, %1, off"
               :: "v"(lds_off), "v"(gptr) : "memory");
}
__device__ __forceinline__ void wait_async0() {
  asm volatile("s_wait_asynccnt 0x0" ::: "memory");
}
__device__ __forceinline__ unsigned lds_off_of(const void* p) {
  return (unsigned)(unsigned long long)p;   // low 32 bits are the LDS offset
}

// Load a 16x32 bf16 fragment (A-layout; also valid as the B operand when B is
// row-major over K, i.e. C = A * B^T) from row-major storage, leading dim ld.
// Lane L(0..15): row L, K={0..7,16..23}; lane L+16: row L, K={8..15,24..31}.
__device__ __forceinline__ v16bf load_frag_rm(const __bf16* __restrict__ tile, int ld) {
  const int lane = lane_id();
  const __bf16* p = tile + (long)(lane & 15) * ld + ((lane >> 4) << 3);
  v8bf lo = *(const v8bf*)(p);
  v8bf hi = *(const v8bf*)(p + 16);
  v16bf r;
#pragma unroll
  for (int i = 0; i < 8; ++i) { r[i] = lo[i]; r[i + 8] = hi[i]; }
  return r;
}

__device__ __forceinline__ v8f wmma_bf16(v16bf a, v16bf b, v8f c) {
  return __builtin_amdgcn_wmma_f32_16x16x32_bf16(false, a, false, b, (short)0, c,
                                                 false, false);
}

// ---------------------------------------------------------------------------
// fp32 -> bf16 conversion (grid-stride)
// ---------------------------------------------------------------------------
__global__ void f32_to_bf16(const float* __restrict__ s, __bf16* __restrict__ d, long n) {
  long i = (long)blockIdx.x * blockDim.x + threadIdx.x;
  long st = (long)gridDim.x * blockDim.x;
  for (; i < n; i += st) d[i] = (__bf16)s[i];
}

// ---------------------------------------------------------------------------
// Generic C = A * B^T GEMM (bf16 in, f32 accumulate).
//   A: [M x K] row-major (+ batch stride sa)
//   B: [N x K] row-major (+ batch stride sb)
//   flags bit0 -> f32 out, else bf16; bit1 -> transposed store (V^T[b][d][t])
// Workgroup: 4 waves; tile BM=128 (32 rows/wave), BN=64.
// Each k-step: 2 A frags (global, L2-hit) x 4 B frags (LDS) -> 8 WMMAs/wave.
// B tile (64x32 bf16, 4KB) DMA'd into LDS with async-to-LDS.
// ---------------------------------------------------------------------------
__global__ __launch_bounds__(128) void gemm_abt(
    const __bf16* __restrict__ A, long sa,
    const __bf16* __restrict__ B, long sb,
    void* __restrict__ Cv, long sc,
    int M, int N, int K, int tdim, int flags)
{
  __shared__ __bf16 sB[64 * 32];
  const int wave = (int)(threadIdx.x >> 5);
  const int lane = lane_id();
  const int bat  = (int)blockIdx.z;
  const long n0  = (long)blockIdx.x * 64;
  const long m0  = (long)blockIdx.y * 128 + wave * 32;

  const __bf16* Ab = A + (long)bat * sa;
  const __bf16* Bb = B + (long)bat * sb;

  // each thread DMAs two 16B chunks of the 64x32 tile
  const int idx0 = (int)threadIdx.x * 8;            // element index, multiple of 8
  const int r0 = idx0 >> 5,            c0 = idx0 & 31;
  const int r1 = (idx0 + 1024) >> 5,   c1 = (idx0 + 1024) & 31;
  const unsigned l0 = lds_off_of(&sB[idx0]);
  const unsigned l1 = lds_off_of(&sB[idx0 + 1024]);

  v8f acc[2][4] = {};

  for (int k0 = 0; k0 < K; k0 += 32) {
    __syncthreads();   // prior-iteration readers done before overwriting sB
    async_copy_b128(l0, Bb + (n0 + r0) * (long)K + k0 + c0);
    async_copy_b128(l1, Bb + (n0 + r1) * (long)K + k0 + c1);
    wait_async0();
    __syncthreads();   // all waves' DMAs landed

    v16bf fa0 = load_frag_rm(Ab + (m0 +  0) * (long)K + k0, K);
    v16bf fa1 = load_frag_rm(Ab + (m0 + 16) * (long)K + k0, K);
    if (k0 + 32 < K) {
      __builtin_prefetch(Ab + m0 * (long)K + k0 + 32, 0, 0);   // global_prefetch_b8
      __builtin_prefetch(Bb + n0 * (long)K + k0 + 32, 0, 0);
    }
#pragma unroll
    for (int j = 0; j < 4; ++j) {
      v16bf fb = load_frag_rm(&sB[j * 16 * 32], 32);
      acc[0][j] = wmma_bf16(fa0, fb, acc[0][j]);
      acc[1][j] = wmma_bf16(fa1, fb, acc[1][j]);
    }
  }

  // Epilogue. C fragment layout: VGPR r -> m = r + 8*(lane>=16), n = lane&15.
#pragma unroll
  for (int h = 0; h < 2; ++h) {
#pragma unroll
    for (int j = 0; j < 4; ++j) {
#pragma unroll
      for (int r = 0; r < 8; ++r) {
        int m = (int)m0 + h * 16 + r + ((lane >> 4) << 3);
        int n = (int)n0 + j * 16 + (lane & 15);
        float v = acc[h][j][r];
        if (flags & 2) {            // transposed store: V^T[b][n][t]
          int b2 = m / tdim, t = m % tdim;
          ((__bf16*)Cv)[((long)b2 * N + n) * tdim + t] = (__bf16)v;
        } else if (flags & 1) {     // f32 output
          ((float*)Cv)[(long)bat * sc + (long)m * N + n] = v;
        } else {                    // bf16 output
          ((__bf16*)Cv)[(long)bat * sc + (long)m * N + n] = (__bf16)v;
        }
      }
    }
  }
}

// ---------------------------------------------------------------------------
// Fused causal scores + softmax: P[b][m][:] = softmax(mask(Q K^T / 32)).
// One workgroup (8 waves, 256 thr) per (16-row query tile, batch).
// Dynamic LDS 160KB: Q tile 16x1024 bf16 (32KB) + score strip 16x2048 f32
// (128KB). Only tiles at/below the diagonal are computed.
// ---------------------------------------------------------------------------
__global__ __launch_bounds__(256) void attn_scores(
    const __bf16* __restrict__ Q,
    const __bf16* __restrict__ Kmat,
    __bf16* __restrict__ P)
{
  extern __shared__ char smem[];
  __bf16* sQ = (__bf16*)smem;                          // [16][DMODEL]
  float*  sS = (float*)(smem + 16 * DMODEL * 2);       // [16][TSEQ]

  const int m0   = (int)blockIdx.x * 16;
  const int b    = (int)blockIdx.y;
  const int wave = (int)(threadIdx.x >> 5);
  const int lane = lane_id();

  // Async-DMA the Q tile (16 rows contiguous in global): 32KB, 8x16B/thread.
  const __bf16* Qb = Q + ((long)b * TSEQ + m0) * DMODEL;
  for (int idx = (int)threadIdx.x * 8; idx < 16 * DMODEL; idx += 256 * 8)
    async_copy_b128(lds_off_of(&sQ[idx]), Qb + idx);
  wait_async0();
  __syncthreads();

  const int lastTile = m0 / 16;           // causal: key tiles 0..lastTile
  const float scale = 0.03125f;           // 1/sqrt(1024)

  for (int jt = wave; jt <= lastTile; jt += 8) {
    const int n0 = jt * 16;
    v8f acc = {};
    const __bf16* Kb = Kmat + ((long)b * TSEQ + n0) * DMODEL;
    for (int k0 = 0; k0 < DMODEL; k0 += 32) {
      v16bf fa = load_frag_rm(sQ + k0, DMODEL);
      v16bf fb = load_frag_rm(Kb + k0, DMODEL);
      acc = wmma_bf16(fa, fb, acc);
    }
#pragma unroll
    for (int r = 0; r < 8; ++r) {
      int mrow = r + ((lane >> 4) << 3);
      int gm = m0 + mrow;
      int gn = n0 + (lane & 15);
      sS[mrow * TSEQ + gn] = (gn <= gm) ? acc[r] * scale : -3.0e38f;
    }
  }
  __syncthreads();

  // Row softmax: 16 threads (half-wave) per row.
  const int row = (int)(threadIdx.x >> 4);
  const int l16 = (int)(threadIdx.x & 15);
  const int gm  = m0 + row;
  const int L   = gm + 1;                 // valid columns [0, L)

  float mx = -3.0e38f;
  for (int c = l16; c < L; c += 16) mx = fmaxf(mx, sS[row * TSEQ + c]);
#pragma unroll
  for (int o = 8; o > 0; o >>= 1) mx = fmaxf(mx, __shfl_xor(mx, o, 16));

  float sum = 0.f;
  for (int c = l16; c < L; c += 16) {
    float e = __expf(sS[row * TSEQ + c] - mx);
    sS[row * TSEQ + c] = e;
    sum += e;
  }
#pragma unroll
  for (int o = 8; o > 0; o >>= 1) sum += __shfl_xor(sum, o, 16);
  float inv = 1.f / sum;

  __bf16* Pr = P + ((long)b * TSEQ + gm) * TSEQ;
  for (int c = l16; c < TSEQ; c += 16)
    Pr[c] = (c < L) ? (__bf16)(sS[row * TSEQ + c] * inv) : (__bf16)0.f;
}

// ---------------------------------------------------------------------------
// Host-side orchestration
// ---------------------------------------------------------------------------
extern "C" void kernel_launch(void* const* d_in, const int* in_sizes, int n_in,
                              void* d_out, int out_size, void* d_ws, size_t ws_size,
                              hipStream_t stream) {
  const float* x  = (const float*)d_in[0];
  const float* Wq = (const float*)d_in[1];
  const float* Wk = (const float*)d_in[2];
  const float* Wv = (const float*)d_in[3];
  const float* Wo = (const float*)d_in[4];

  // Workspace layout (bf16)
  char* w = (char*)d_ws;
  __bf16* xb  = (__bf16*)w; w += (size_t)BT * DMODEL * 2;           // 16 MB
  __bf16* Wqb = (__bf16*)w; w += (size_t)DMODEL * DMODEL * 2;       //  2 MB
  __bf16* Wkb = (__bf16*)w; w += (size_t)DMODEL * DMODEL * 2;
  __bf16* Wvb = (__bf16*)w; w += (size_t)DMODEL * DMODEL * 2;
  __bf16* Wob = (__bf16*)w; w += (size_t)DMODEL * DMODEL * 2;
  __bf16* qb  = (__bf16*)w; w += (size_t)BT * DMODEL * 2;           // 16 MB
  __bf16* kb  = (__bf16*)w; w += (size_t)BT * DMODEL * 2;           // 16 MB
  __bf16* vtb = (__bf16*)w; w += (size_t)BATCH * DMODEL * TSEQ * 2; // V^T, 16 MB
  __bf16* Pb  = (__bf16*)w; w += (size_t)BATCH * TSEQ * TSEQ * 2;   // 32 MB
  __bf16* ob  = (__bf16*)w; w += (size_t)BT * DMODEL * 2;           // 16 MB

  // 1) convert fp32 -> bf16
  f32_to_bf16<<<4096, 256, 0, stream>>>(x,  xb,  (long)BT * DMODEL);
  f32_to_bf16<<<1024, 256, 0, stream>>>(Wq, Wqb, (long)DMODEL * DMODEL);
  f32_to_bf16<<<1024, 256, 0, stream>>>(Wk, Wkb, (long)DMODEL * DMODEL);
  f32_to_bf16<<<1024, 256, 0, stream>>>(Wv, Wvb, (long)DMODEL * DMODEL);
  f32_to_bf16<<<1024, 256, 0, stream>>>(Wo, Wob, (long)DMODEL * DMODEL);

  const dim3 blk(128);
  // 2) projections: q = x Wq^T, k = x Wk^T, v^T (transposed store)
  dim3 gproj(DMODEL / 64, BT / 128, 1);
  gemm_abt<<<gproj, blk, 0, stream>>>(xb, 0, Wqb, 0, qb, 0,
                                      BT, DMODEL, DMODEL, TSEQ, /*bf16*/0);
  gemm_abt<<<gproj, blk, 0, stream>>>(xb, 0, Wkb, 0, kb, 0,
                                      BT, DMODEL, DMODEL, TSEQ, 0);
  gemm_abt<<<gproj, blk, 0, stream>>>(xb, 0, Wvb, 0, vtb, 0,
                                      BT, DMODEL, DMODEL, TSEQ, /*transpose*/2);

  // 3) fused causal scores + softmax -> P
  size_t smemBytes = (size_t)16 * DMODEL * 2 + (size_t)16 * TSEQ * 4; // 160 KB
  attn_scores<<<dim3(TSEQ / 16, BATCH), 256, smemBytes, stream>>>(qb, kb, Pb);

  // 4) out = P @ V = P @ (V^T)^T, batched over B
  dim3 gpv(DMODEL / 64, TSEQ / 128, BATCH);
  gemm_abt<<<gpv, blk, 0, stream>>>(Pb,  (long)TSEQ * TSEQ,
                                    vtb, (long)DMODEL * TSEQ,
                                    ob,  (long)TSEQ * DMODEL,
                                    TSEQ, DMODEL, TSEQ, TSEQ, 0);

  // 5) final projection: d_out(f32) = out @ Wo^T
  gemm_abt<<<gproj, blk, 0, stream>>>(ob, 0, Wob, 0, d_out, 0,
                                      BT, DMODEL, DMODEL, TSEQ, /*f32*/1);
}
